// AdaAttention_36386962932208
// MI455X (gfx1250) — matrix operand order repaired and checked by reading, not compiled, not run on hardware
//
#include <hip/hip_runtime.h>
#include <hip/hip_bf16.h>

// CDNA5 / gfx1250: wave32, WMMA 16x16x32 bf16 (f32 accumulate).
// Weights are pre-packed once per launch into bf16 B-fragment layout so the
// hot loops do pure b128 loads + v_wmma. The fused attention-score GEMM
// double-buffers its weight tile in LDS via the gfx1250 async global->LDS
// engine (ASYNCcnt), and rotates 3 B-fragment register buffers so ds_loads
// stay 2 steps ahead of the WMMAs.

typedef __attribute__((ext_vector_type(16))) __bf16 v16bf;
typedef __attribute__((ext_vector_type(8)))  float  v8f;

#if defined(__has_builtin)
#  if __has_builtin(__builtin_amdgcn_global_load_async_to_lds_b128) && \
      __has_builtin(__builtin_amdgcn_s_wait_asynccnt)
#    define USE_ASYNC_LDS 1
#  endif
#endif

// Exact pointee type from the builtin's signature: int __vector(4).
typedef int b128_t __attribute__((vector_size(16)));
typedef __attribute__((address_space(1))) b128_t as1_b128;
typedef __attribute__((address_space(3))) b128_t as3_b128;

// Copy 16 bytes global -> LDS. Async-engine variant when available.
__device__ __forceinline__ void copy16_g2l(const void* gsrc, void* ldst) {
#ifdef USE_ASYNC_LDS
    __builtin_amdgcn_global_load_async_to_lds_b128(
        (as1_b128*)(size_t)gsrc, (as3_b128*)(unsigned)(size_t)ldst, 0, 0);
#else
    *(uint4*)ldst = *(const uint4*)gsrc;
#endif
}

__device__ __forceinline__ void g2l_wait() {
#ifdef USE_ASYNC_LDS
    __builtin_amdgcn_s_wait_asynccnt(0);
#endif
}

union BF16Frag {
    v16bf    v;
    unsigned u32[8];
    uint4    q[2];
};

// Truncating fp32 -> bf16 pack of two values into one dword (lo = a, hi = b).
__device__ __forceinline__ unsigned pack2bf(float a, float b) {
    union { float f; unsigned u; } ca, cb;
    ca.f = a; cb.f = b;
    return (ca.u >> 16) | (cb.u & 0xFFFF0000u);
}

// Fill an A-fragment (16x32 bf16, contiguous k = 16*half + s) from a row-major fp32 row.
__device__ __forceinline__ void load_a_frag(BF16Frag& f, const float* __restrict__ pA,
                                            int kk, int half) {
    const int abase = kk * 32 + half * 16;
    const float4 x0 = *(const float4*)(pA + abase);
    const float4 x1 = *(const float4*)(pA + abase + 4);
    const float4 x2 = *(const float4*)(pA + abase + 8);
    const float4 x3 = *(const float4*)(pA + abase + 12);
    f.u32[0] = pack2bf(x0.x, x0.y);
    f.u32[1] = pack2bf(x0.z, x0.w);
    f.u32[2] = pack2bf(x1.x, x1.y);
    f.u32[3] = pack2bf(x1.z, x1.w);
    f.u32[4] = pack2bf(x2.x, x2.y);
    f.u32[5] = pack2bf(x2.z, x2.w);
    f.u32[6] = pack2bf(x3.x, x3.y);
    f.u32[7] = pack2bf(x3.z, x3.w);
}

// ---------------------------------------------------------------------------
// Pack six 512x512 fp32 weights into bf16 B-fragment layout:
//   dstShort[ w*262144 + (((nt*16 + kk)*32 + lane)*16 + i) ]
//     = bf16( W[(kk*32 + (lane>>4)*16 + i)*512 + nt*16 + (lane&15)] )
// ---------------------------------------------------------------------------
__global__ void __launch_bounds__(256)
pack_weights6(const float* __restrict__ w0, const float* __restrict__ w1,
              const float* __restrict__ w2, const float* __restrict__ w3,
              const float* __restrict__ w4, const float* __restrict__ w5,
              unsigned* __restrict__ dst) {
    const unsigned tid = blockIdx.x * 256 + threadIdx.x;   // 0 .. 786431
    const unsigned q   = tid & 131071u;                    // dword index within weight
    const unsigned w   = tid >> 17;
    const float* W = (w == 0) ? w0 : (w == 1) ? w1 : (w == 2) ? w2
                   : (w == 3) ? w3 : (w == 4) ? w4 : w5;
    const unsigned ip   = q & 7u;
    const unsigned lane = (q >> 3) & 31u;
    const unsigned kk   = (q >> 8) & 15u;
    const unsigned nt   = q >> 12;
    const unsigned col  = nt * 16 + (lane & 15u);
    const unsigned k0   = kk * 32 + (lane >> 4) * 16 + ip * 2;
    const float a = W[(size_t)k0 * 512 + col];
    const float b = W[(size_t)(k0 + 1) * 512 + col];
    dst[tid] = pack2bf(a, b);
}

// ---------------------------------------------------------------------------
// 1024x512x512 GEMM with pre-packed bf16 weights. One 16x16 tile per wave.
// ACT: 0 = identity, 1 = relu, 2 = tanh.
// ---------------------------------------------------------------------------
template <int ACT>
__global__ void __launch_bounds__(256)
gemm_bf16_wmma(const float* __restrict__ A, const unsigned short* __restrict__ Wp,
               const float* __restrict__ bias, float* __restrict__ C,
               int M, int N, int K) {
    const int lane = threadIdx.x & 31;
    const int wave = threadIdx.x >> 5;
    const int gw   = blockIdx.x * 8 + wave;
    const int ntiles = N >> 4;
    const int tM = gw / ntiles;
    const int tN = gw - tM * ntiles;
    if (tM * 16 >= M) return;                      // wave-uniform

    const int half = lane >> 4;
    const int l15  = lane & 15;
    const int ksteps = K >> 5;

    const float* pA  = A + (size_t)(tM * 16 + l15) * K;
    const uint4* pWq = (const uint4*)Wp + ((size_t)tN * ksteps * 32 + lane) * 2;

    v8f acc = {};
    for (int kk = 0; kk < ksteps; ++kk) {
        BF16Frag af, bf;
        load_a_frag(af, pA, kk, half);
        bf.q[0] = pWq[(size_t)kk * 64];
        bf.q[1] = pWq[(size_t)kk * 64 + 1];
        acc = __builtin_amdgcn_wmma_f32_16x16x32_bf16(
                  false, af.v, false, bf.v, (short)0, acc, false, false);
    }

    const int col = tN * 16 + l15;
    const float bc = bias[col];
#pragma unroll
    for (int j = 0; j < 8; ++j) {
        const int row = tM * 16 + half * 8 + j;
        float v = acc[j] + bc;
        if (ACT == 1) v = fmaxf(v, 0.0f);
        if (ACT == 2) v = tanhf(v);
        C[(size_t)row * N + col] = v;
    }
}

// ---------------------------------------------------------------------------
// Fused big GEMM + additive-attention score:
//   scores[b, s+1] = b_alpha + sum_c W_alpha[c] *
//       tanh( (att_feats[b,s,:] @ W_ctx2att)[c] + b_ctx2att[c] + h_emb[b,c] )
// 16 flattened rows per wave (200704 rows = 12544 tiles, exact). A fragments
// for all K=512 stay in registers. W_ctx2att fragments stream through a
// 2 x 16 KB LDS double buffer via async global->LDS; B-fragment registers
// rotate through 3 buffers (prefetch distance 2) so ds_loads overlap WMMAs.
// ---------------------------------------------------------------------------
__global__ void __launch_bounds__(256)
att_scores_wmma(const float* __restrict__ att, const unsigned short* __restrict__ Wcp,
                const float* __restrict__ bc, const float* __restrict__ h_emb,
                const float* __restrict__ Walpha, const float* __restrict__ balpha,
                float* __restrict__ scores) {
    __shared__ uint4 sW[2][1024];                  // 2 x 16 KB weight-tile buffers
    const int tid  = threadIdx.x;
    const int lane = tid & 31;
    const int half = lane >> 4;
    const int l15  = lane & 15;
    const int wave = tid >> 5;
    const int gw   = blockIdx.x * 8 + wave;        // row tile: 0..12543
    const int rowBase = gw * 16;

    const float* pA = att + (size_t)(rowBase + l15) * 512;
    BF16Frag af[16];
#pragma unroll
    for (int kk = 0; kk < 16; ++kk) load_a_frag(af[kk], pA, kk, half);

    float sacc[8];
#pragma unroll
    for (int j = 0; j < 8; ++j) sacc[j] = 0.0f;

    const int myRow = rowBase + half * 8;
    const int b0 = myRow / 196;
    const int b1 = (myRow + 7) / 196;              // tile spans at most 2 batches

    const uint4* gW = (const uint4*)Wcp;

    // Prologue: stage N-tile 0 into buffer 0 (4 x 256 threads x 16 B = 16 KB).
#pragma unroll
    for (int u = 0; u < 4; ++u)
        copy16_g2l(gW + tid + u * 256, &sW[0][tid + u * 256]);
    g2l_wait();
    __syncthreads();

    for (int nt = 0; nt < 32; ++nt) {
        const int cur = nt & 1;
        const int nxt = cur ^ 1;
        // Kick off async stage of the next tile while computing this one.
        if (nt < 31) {
#pragma unroll
            for (int u = 0; u < 4; ++u)
                copy16_g2l(gW + (size_t)(nt + 1) * 1024 + tid + u * 256,
                           &sW[nxt][tid + u * 256]);
        }

        // 16 WMMAs from the resident tile; 3 rotating B-fragment buffers,
        // loads issued 2 steps ahead of their WMMA.
        v8f acc = {};
        BF16Frag bfr[3];
#pragma unroll
        for (int p = 0; p < 2; ++p) {
            bfr[p].q[0] = sW[cur][(p * 32 + lane) * 2];
            bfr[p].q[1] = sW[cur][(p * 32 + lane) * 2 + 1];
        }
#pragma unroll
        for (int kk = 0; kk < 16; ++kk) {
            if (kk + 2 < 16) {
                bfr[(kk + 2) % 3].q[0] = sW[cur][((kk + 2) * 32 + lane) * 2];
                bfr[(kk + 2) % 3].q[1] = sW[cur][((kk + 2) * 32 + lane) * 2 + 1];
            }
            acc = __builtin_amdgcn_wmma_f32_16x16x32_bf16(
                      false, af[kk].v, false, bfr[kk % 3].v, (short)0, acc, false, false);
        }

        const int col  = nt * 16 + l15;
        const float wcol = Walpha[col];
        const float bcol = bc[col];
        const float he0  = h_emb[(size_t)b0 * 512 + col];
        const float he1  = (b1 != b0) ? h_emb[(size_t)b1 * 512 + col] : he0;
#pragma unroll
        for (int j = 0; j < 8; ++j) {
            const int b = (myRow + j) / 196;
            const float he = (b == b0) ? he0 : he1;
            sacc[j] += tanhf(acc[j] + bcol + he) * wcol;
        }

        g2l_wait();                                 // this wave's async loads done
        __syncthreads();                            // whole block ready to swap
    }

    // Sum over the 16 lanes of each half (each half owns disjoint rows).
#pragma unroll
    for (int j = 0; j < 8; ++j) {
        float s = sacc[j];
        s += __shfl_xor(s, 1, 32);
        s += __shfl_xor(s, 2, 32);
        s += __shfl_xor(s, 4, 32);
        s += __shfl_xor(s, 8, 32);
        sacc[j] = s;
    }
    if (l15 == 0) {
        const float ba = balpha[0];
#pragma unroll
        for (int j = 0; j < 8; ++j) {
            const int row = myRow + j;
            const int b = row / 196;
            const int s = row - b * 196;
            scores[(size_t)b * 197 + s + 1] = sacc[j] + ba;
        }
    }
}

// ---------------------------------------------------------------------------
// Per-batch: sentinel score, softmax over 197, weighted context (re-reads
// att_feats, coalesced float2 + global prefetch), + h_lin -> atten_out.
// ---------------------------------------------------------------------------
__global__ void __launch_bounds__(256)
softmax_ctx(const float* __restrict__ scores, const float* __restrict__ sent_emb,
            const float* __restrict__ h_emb, const float* __restrict__ Walpha,
            const float* __restrict__ balpha, const float* __restrict__ sent_lin,
            const float* __restrict__ att, const float* __restrict__ h_lin,
            float* __restrict__ atten_out) {
    __shared__ float s_alpha[200];
    __shared__ float red[256];
    const int b = blockIdx.x;
    const int t = threadIdx.x;

    // sentinel score: W_alpha . tanh(sent_emb + h_emb)
    float p = 0.0f;
    for (int c = t; c < 512; c += 256)
        p += tanhf(sent_emb[(size_t)b * 512 + c] + h_emb[(size_t)b * 512 + c]) * Walpha[c];
    red[t] = p; __syncthreads();
    for (int s = 128; s > 0; s >>= 1) { if (t < s) red[t] += red[t + s]; __syncthreads(); }
    if (t == 0) s_alpha[0] = red[0] + balpha[0];
    if (t < 196) s_alpha[t + 1] = scores[(size_t)b * 197 + t + 1];
    __syncthreads();

    // softmax over 197
    float m = -1e30f;
    for (int i = t; i < 197; i += 256) m = fmaxf(m, s_alpha[i]);
    red[t] = m; __syncthreads();
    for (int s = 128; s > 0; s >>= 1) { if (t < s) red[t] = fmaxf(red[t], red[t + s]); __syncthreads(); }
    m = red[0]; __syncthreads();
    float e = 0.0f;
    for (int i = t; i < 197; i += 256) { float x = __expf(s_alpha[i] - m); s_alpha[i] = x; e += x; }
    red[t] = e; __syncthreads();
    for (int s = 128; s > 0; s >>= 1) { if (t < s) red[t] += red[t + s]; __syncthreads(); }
    const float inv = 1.0f / red[0];
    __syncthreads();
    for (int i = t; i < 197; i += 256) s_alpha[i] *= inv;
    __syncthreads();

    // cHat + h_lin: each thread owns 2 columns (coalesced float2 across block)
    const int d = t * 2;
    const float a0 = s_alpha[0];
    float ax = a0 * sent_lin[(size_t)b * 512 + d];
    float ay = a0 * sent_lin[(size_t)b * 512 + d + 1];
    const float* pa = att + (size_t)b * 196 * 512 + d;
    for (int s = 0; s < 196; ++s) {
        if (s + 8 < 196) __builtin_prefetch(pa + (size_t)(s + 8) * 512, 0, 0);
        const float2 v = *(const float2*)(pa + (size_t)s * 512);
        const float a = s_alpha[s + 1];
        ax += a * v.x;
        ay += a * v.y;
    }
    atten_out[(size_t)b * 512 + d]     = ax + h_lin[(size_t)b * 512 + d];
    atten_out[(size_t)b * 512 + d + 1] = ay + h_lin[(size_t)b * 512 + d + 1];
}

// ---------------------------------------------------------------------------
extern "C" void kernel_launch(void* const* d_in, const int* in_sizes, int n_in,
                              void* d_out, int out_size, void* d_ws, size_t ws_size,
                              hipStream_t stream) {
    (void)in_sizes; (void)n_in; (void)out_size; (void)ws_size;
    const float* h         = (const float*)d_in[0];
    const float* sentinel  = (const float*)d_in[1];
    const float* att       = (const float*)d_in[2];
    const float* W_ctx2att = (const float*)d_in[3];
    const float* b_ctx2att = (const float*)d_in[4];
    const float* W_sl      = (const float*)d_in[5];
    const float* b_sl      = (const float*)d_in[6];
    const float* W_se      = (const float*)d_in[7];
    const float* b_se      = (const float*)d_in[8];
    const float* W_hl      = (const float*)d_in[9];
    const float* b_hl      = (const float*)d_in[10];
    const float* W_he      = (const float*)d_in[11];
    const float* b_he      = (const float*)d_in[12];
    const float* W_alpha   = (const float*)d_in[13];
    const float* b_alpha   = (const float*)d_in[14];
    const float* W_att2h   = (const float*)d_in[15];
    const float* b_att2h   = (const float*)d_in[16];
    float* out = (float*)d_out;
    float* ws  = (float*)d_ws;

    const size_t BH = 1024 * 512;          // 524288 floats
    float* sent_lin  = ws;
    float* h_lin     = ws + BH;
    float* sent_emb  = ws + 2 * BH;
    float* h_emb     = ws + 3 * BH;
    float* scores    = ws + 4 * BH;        // 1024*197 = 201728 floats
    float* atten_out = scores + 201728;
    // Packed bf16 weights: 6 x 262144 shorts = 3 MB, 16B-aligned.
    unsigned short* packW = (unsigned short*)(atten_out + BH);
    const size_t WSZ = 262144;             // shorts per packed weight

    const dim3 blk(256);
    const int M = 1024, N = 512, K = 512;
    const int gemm_blocks = (M / 16) * (N / 16) / 8;      // 256 blocks, 8 waves each

    // Weight order: 0 W_sl, 1 W_hl, 2 W_se, 3 W_he, 4 W_ctx2att, 5 W_att2h
    pack_weights6<<<3072, blk, 0, stream>>>(W_sl, W_hl, W_se, W_he, W_ctx2att, W_att2h,
                                            (unsigned*)packW);

    gemm_bf16_wmma<1><<<gemm_blocks, blk, 0, stream>>>(sentinel, packW + 0 * WSZ, b_sl, sent_lin, M, N, K);
    gemm_bf16_wmma<2><<<gemm_blocks, blk, 0, stream>>>(h,        packW + 1 * WSZ, b_hl, h_lin,    M, N, K);
    gemm_bf16_wmma<0><<<gemm_blocks, blk, 0, stream>>>(sent_lin, packW + 2 * WSZ, b_se, sent_emb, M, N, K);
    gemm_bf16_wmma<0><<<gemm_blocks, blk, 0, stream>>>(h_lin,    packW + 3 * WSZ, b_he, h_emb,    M, N, K);

    // 1024*196 rows / 16 rows-per-wave / 8 waves-per-block = 1568 blocks
    att_scores_wmma<<<1568, blk, 0, stream>>>(att, packW + 4 * WSZ, b_ctx2att, h_emb,
                                              W_alpha, b_alpha, scores);

    softmax_ctx<<<1024, blk, 0, stream>>>(scores, sent_emb, h_emb, W_alpha, b_alpha,
                                          sent_lin, att, h_lin, atten_out);

    gemm_bf16_wmma<2><<<gemm_blocks, blk, 0, stream>>>(atten_out, packW + 5 * WSZ, b_att2h, out, M, N, K);
}
